// Attention_88656714925044
// MI455X (gfx1250) — matrix operand order repaired and checked by reading, not compiled, not run on hardware
//
#include <hip/hip_runtime.h>
#include <hip/hip_bf16.h>

#define DIM   1024
#define SEQ   2048
#define BATCH 2
#define NHEAD 16
#define HD    64
#define TOKS  (BATCH*SEQ)   // 4096

typedef __attribute__((ext_vector_type(16))) __bf16 v16bf;
typedef __attribute__((ext_vector_type(8)))  __bf16 v8bf;
typedef __attribute__((ext_vector_type(8)))  float  v8f;
typedef __attribute__((ext_vector_type(4)))  int    v4i;

// ---------------- gfx1250 async global->LDS path (guarded) ---------------------

#if __has_builtin(__builtin_amdgcn_global_load_async_to_lds_b128) && \
    __has_builtin(__builtin_amdgcn_s_wait_asynccnt)
#define USE_ASYNC_LDS 1
#define ASYNC_WAIT(n) __builtin_amdgcn_s_wait_asynccnt(n)
#else
#define USE_ASYNC_LDS 0
#define ASYNC_WAIT(n)
#endif

typedef __attribute__((address_space(1))) v4i* gptr_b128;  // global src
typedef __attribute__((address_space(3))) v4i* lptr_b128;  // LDS dst

// copy 16 bytes (8 bf16) per lane from global to LDS
static __device__ __forceinline__ void async_copy16(const __bf16* g, __bf16* l) {
#if USE_ASYNC_LDS
  __builtin_amdgcn_global_load_async_to_lds_b128((gptr_b128)g, (lptr_b128)l, 0, 0);
#else
  *reinterpret_cast<v8bf*>(l) = *reinterpret_cast<const v8bf*>(g);
#endif
}

// ---------------- WMMA fragment helpers (ISA 7.12.2 layouts, wave32) -----------

// A fragment: 16(M) x 32(K) bf16, element (m,k) at p[m*ld + k].
static __device__ __forceinline__ v16bf load_a_bf16(const __bf16* p, int ld) {
  const int l  = threadIdx.x & 31;
  const int m  = l & 15;
  const int hi = l >> 4;
  const __bf16* row = p + (size_t)m * ld;
  v8bf c0 = *reinterpret_cast<const v8bf*>(row + hi * 8);
  v8bf c1 = *reinterpret_cast<const v8bf*>(row + 16 + hi * 8);
  v16bf a;
#pragma unroll
  for (int i = 0; i < 8; ++i) { a[i] = c0[i]; a[i + 8] = c1[i]; }
  return a;
}

// B fragment: 32(K) x 16(N) bf16, element (k,n) at p[n*ld + k] (K contiguous).
static __device__ __forceinline__ v16bf load_b_bf16(const __bf16* p, int ld) {
  const int l  = threadIdx.x & 31;
  const int n  = l & 15;
  const int hi = l >> 4;
  const __bf16* col = p + (size_t)n * ld + hi * 8;
  v8bf c0 = *reinterpret_cast<const v8bf*>(col);
  v8bf c1 = *reinterpret_cast<const v8bf*>(col + 16);
  v16bf b;
#pragma unroll
  for (int i = 0; i < 8; ++i) { b[i] = c0[i]; b[i + 8] = c1[i]; }
  return b;
}

static __device__ __forceinline__ v8f wmma_bf16(v16bf a, v16bf b, v8f c) {
  return __builtin_amdgcn_wmma_f32_16x16x32_bf16(false, a, false, b, (short)0, c,
                                                 false, false);
}

// ---------------- precision-conversion kernels ---------------------------------

__global__ void k_cvt_bf16(const float* __restrict__ src, __bf16* __restrict__ dst, int n) {
  int i = blockIdx.x * blockDim.x + threadIdx.x;
  if (i < n) dst[i] = (__bf16)src[i];
}

// dst[c*DIM + r] = (bf16) src[r*DIM + c]  ->  wT[out][in], K contiguous for B frags
__global__ void k_transpose_bf16(const float* __restrict__ src, __bf16* __restrict__ dst) {
  int c = blockIdx.x * 16 + threadIdx.x;
  int r = blockIdx.y * 16 + threadIdx.y;
  dst[(size_t)c * DIM + r] = (__bf16)src[(size_t)r * DIM + c];
}

// ---------------- fused QKV projection ----------------------------------------
// grid (TOKS/64, DIM/64, 3), block 128 (4 waves). Block: 64 rows x 64 cols.
// B strip (64x32 bf16 per k-step) staged in LDS via async copies, double-buffered.
__global__ void k_gemm_qkv(const __bf16* __restrict__ xb,
                           const __bf16* __restrict__ wqT, const __bf16* __restrict__ wkT,
                           const __bf16* __restrict__ wvT,
                           const float* __restrict__ bq, const float* __restrict__ bk,
                           const float* __restrict__ bv,
                           __bf16* __restrict__ Q, __bf16* __restrict__ K,
                           __bf16* __restrict__ Vt) {
  __shared__ __align__(16) __bf16 bsm[2][64 * 32];   // 2 x 4KB B tiles

  const int tid   = threadIdx.x;
  const int wave  = tid >> 5;
  const int tok0  = (blockIdx.x * 4 + wave) * 16;
  const int col0  = blockIdx.y * 64;
  const int which = blockIdx.z;
  const __bf16* wT  = (which == 0) ? wqT : (which == 1) ? wkT : wvT;
  const float* bias = (which == 0) ? bq  : (which == 1) ? bk  : bv;

  auto stage = [&](int buf, int k0) {           // 128 thr * 32B = 4KB tile
    int r = tid >> 1, c = (tid & 1) * 16;
    const __bf16* s = wT + (size_t)(col0 + r) * DIM + k0 + c;
    __bf16* d = &bsm[buf][r * 32 + c];
    async_copy16(s, d);
    async_copy16(s + 8, d + 8);
  };

  v8f acc[4] = {};
  stage(0, 0);
  int cur = 0;
  for (int k0 = 0; k0 < DIM; k0 += 32) {
    if (k0 + 32 < DIM) { stage(cur ^ 1, k0 + 32); ASYNC_WAIT(2); }
    else               { ASYNC_WAIT(0); }
    __syncthreads();
    __builtin_prefetch(xb + (size_t)tok0 * DIM + k0 + 128, 0, 1);
    v16bf a = load_a_bf16(xb + (size_t)tok0 * DIM + k0, DIM);
#pragma unroll
    for (int j = 0; j < 4; ++j) {
      v16bf b = load_b_bf16(&bsm[cur][16 * j * 32], 32);
      acc[j] = wmma_bf16(a, b, acc[j]);
    }
    __syncthreads();
    cur ^= 1;
  }

  const int l = tid & 31, n = l & 15, hi = l >> 4;
#pragma unroll
  for (int j = 0; j < 4; ++j) {
#pragma unroll
    for (int v = 0; v < 8; ++v) {
      int tok = tok0 + v + hi * 8;
      int col = col0 + 16 * j + n;
      float val = acc[j][v] + bias[col];
      int bb = tok >> 11, nn = tok & (SEQ - 1);
      int h = col >> 6, d = col & (HD - 1);
      if (which == 2) {
        Vt[(((size_t)(bb * NHEAD + h)) * HD + d) * SEQ + nn] = (__bf16)val;
      } else {
        __bf16* dstp = (which == 0) ? Q : K;
        dstp[(((size_t)(bb * NHEAD + h)) * SEQ + nn) * HD + d] = (__bf16)val;
      }
    }
  }
}

// ---------------- flash attention ---------------------------------------------
// grid B*H*(SEQ/16)/4 blocks, block 128. All 4 waves of a block share one (b,h),
// so K/V key-blocks (32 keys) are staged once per block in LDS, double-buffered.
__global__ void k_flash_attn(const __bf16* __restrict__ Q, const __bf16* __restrict__ K,
                             const __bf16* __restrict__ Vt, __bf16* __restrict__ O) {
  __shared__ __align__(16) __bf16 kbuf[2][32 * 64];  // [key][d]
  __shared__ __align__(16) __bf16 vbuf[2][64 * 32];  // [d][key]
  __shared__ __align__(16) __bf16 pt[4][16 * 32];    // per-wave P tile

  const int tid  = threadIdx.x;
  const int wave = tid >> 5;
  const int tile = blockIdx.x * 4 + wave;
  const int q0 = (tile & 127) * 16;
  const int bh = tile >> 7;
  const int bb = bh >> 4, h = bh & 15;
  const __bf16* Qh = Q  + (size_t)bh * SEQ * HD;
  const __bf16* Kh = K  + (size_t)bh * SEQ * HD;
  const __bf16* Vh = Vt + (size_t)bh * HD * SEQ;
  const int l = tid & 31, n = l & 15, hi = l >> 4;

  auto stage_kv = [&](int buf, int j0) {   // 4KB K + 4KB V per stage
    { int r = tid >> 2, c = (tid & 3) * 16;            // K: 32 rows x 128B
      const __bf16* s = Kh + (size_t)(j0 + r) * HD + c;
      __bf16* d = &kbuf[buf][r * 64 + c];
      async_copy16(s, d); async_copy16(s + 8, d + 8);
    }
    { int r = tid >> 1, c = (tid & 1) * 16;            // V: 64 rows x 64B
      const __bf16* s = Vh + (size_t)r * SEQ + j0 + c;
      __bf16* d = &vbuf[buf][r * 32 + c];
      async_copy16(s, d); async_copy16(s + 8, d + 8);
    }
  };

  v16bf qa0 = load_a_bf16(Qh + (size_t)q0 * HD + 0,  HD);
  v16bf qa1 = load_a_bf16(Qh + (size_t)q0 * HD + 32, HD);

  v8f o[4] = {};
  float rm[8], rs[8];
#pragma unroll
  for (int v = 0; v < 8; ++v) { rm[v] = -3.0e38f; rs[v] = 0.0f; }

  __bf16* pl = pt[wave];
  const float scale = 0.125f;   // 1/sqrt(64)

  stage_kv(0, 0);
  int cur = 0;
  for (int j0 = 0; j0 < SEQ; j0 += 32) {
    if (j0 + 32 < SEQ) { stage_kv(cur ^ 1, j0 + 32); ASYNC_WAIT(4); }
    else               { ASYNC_WAIT(0); }
    __syncthreads();

    const __bf16* kt = kbuf[cur];
    const __bf16* vt = vbuf[cur];

    // S = Q @ K^T for 32 keys (two 16x16 f32 tiles, Hd split 2x32)
    v8f s0 = {}, s1 = {};
    {
      v16bf b;
      b = load_b_bf16(kt + 0,            64); s0 = wmma_bf16(qa0, b, s0);
      b = load_b_bf16(kt + 32,           64); s0 = wmma_bf16(qa1, b, s0);
      b = load_b_bf16(kt + 16 * 64,      64); s1 = wmma_bf16(qa0, b, s1);
      b = load_b_bf16(kt + 16 * 64 + 32, 64); s1 = wmma_bf16(qa1, b, s1);
    }

    // online softmax (C-frag layout: row = v + 8*hi, col = lane&15)
#pragma unroll
    for (int v = 0; v < 8; ++v) {
      float e0 = s0[v] * scale, e1 = s1[v] * scale;
      float mv = fmaxf(e0, e1);
      mv = fmaxf(mv, __shfl_xor(mv, 1, 32));
      mv = fmaxf(mv, __shfl_xor(mv, 2, 32));
      mv = fmaxf(mv, __shfl_xor(mv, 4, 32));
      mv = fmaxf(mv, __shfl_xor(mv, 8, 32));
      float nm = fmaxf(rm[v], mv);
      float alpha = __expf(rm[v] - nm);
      rm[v] = nm;
      float p0 = __expf(e0 - nm);
      float p1 = __expf(e1 - nm);
      float ps = p0 + p1;
      ps += __shfl_xor(ps, 1, 32);
      ps += __shfl_xor(ps, 2, 32);
      ps += __shfl_xor(ps, 4, 32);
      ps += __shfl_xor(ps, 8, 32);
      rs[v] = rs[v] * alpha + ps;
#pragma unroll
      for (int dc = 0; dc < 4; ++dc) o[dc][v] *= alpha;
      pl[(v + hi * 8) * 32 + n]      = (__bf16)p0;   // C-layout -> row-major P
      pl[(v + hi * 8) * 32 + 16 + n] = (__bf16)p1;
    }

    v16bf pa = load_a_bf16(pl, 32);       // P as A fragment (16x32)
#pragma unroll
    for (int dc = 0; dc < 4; ++dc) {      // O += P @ V
      v16bf vbf = load_b_bf16(vt + dc * 16 * 32, 32);
      o[dc] = wmma_bf16(pa, vbf, o[dc]);
    }
    __syncthreads();
    cur ^= 1;
  }

  // normalize, write O [B,N,D] bf16 for the output projection
#pragma unroll
  for (int v = 0; v < 8; ++v) {
    float inv = 1.0f / rs[v];
    int row = q0 + v + hi * 8;
#pragma unroll
    for (int dc = 0; dc < 4; ++dc) {
      O[((size_t)(bb * SEQ + row)) * DIM + h * HD + dc * 16 + n] =
          (__bf16)(o[dc][v] * inv);
    }
  }
}

// ---------------- output projection: d_out = O @ wo + bo (f32 out) -------------
__global__ void k_gemm_out(const __bf16* __restrict__ Ob, const __bf16* __restrict__ woT,
                           const float* __restrict__ bo, float* __restrict__ out) {
  __shared__ __align__(16) __bf16 bsm[2][64 * 32];

  const int tid  = threadIdx.x;
  const int wave = tid >> 5;
  const int tok0 = (blockIdx.x * 4 + wave) * 16;
  const int col0 = blockIdx.y * 64;

  auto stage = [&](int buf, int k0) {
    int r = tid >> 1, c = (tid & 1) * 16;
    const __bf16* s = woT + (size_t)(col0 + r) * DIM + k0 + c;
    __bf16* d = &bsm[buf][r * 32 + c];
    async_copy16(s, d);
    async_copy16(s + 8, d + 8);
  };

  v8f acc[4] = {};
  stage(0, 0);
  int cur = 0;
  for (int k0 = 0; k0 < DIM; k0 += 32) {
    if (k0 + 32 < DIM) { stage(cur ^ 1, k0 + 32); ASYNC_WAIT(2); }
    else               { ASYNC_WAIT(0); }
    __syncthreads();
    __builtin_prefetch(Ob + (size_t)tok0 * DIM + k0 + 128, 0, 1);
    v16bf a = load_a_bf16(Ob + (size_t)tok0 * DIM + k0, DIM);
#pragma unroll
    for (int j = 0; j < 4; ++j) {
      v16bf b = load_b_bf16(&bsm[cur][16 * j * 32], 32);
      acc[j] = wmma_bf16(a, b, acc[j]);
    }
    __syncthreads();
    cur ^= 1;
  }

  const int l = tid & 31, n = l & 15, hi = l >> 4;
#pragma unroll
  for (int j = 0; j < 4; ++j)
#pragma unroll
    for (int v = 0; v < 8; ++v) {
      int tok = tok0 + v + hi * 8;
      int col = col0 + 16 * j + n;
      out[(size_t)tok * DIM + col] = acc[j][v] + bo[col];
    }
}

// ---------------- host-side launcher ------------------------------------------

extern "C" void kernel_launch(void* const* d_in, const int* in_sizes, int n_in,
                              void* d_out, int out_size, void* d_ws, size_t ws_size,
                              hipStream_t stream) {
  const float* x  = (const float*)d_in[0];
  const float* wq = (const float*)d_in[1];
  const float* bq = (const float*)d_in[2];
  const float* wk = (const float*)d_in[3];
  const float* bk = (const float*)d_in[4];
  const float* wv = (const float*)d_in[5];
  const float* bv = (const float*)d_in[6];
  const float* wo = (const float*)d_in[7];
  const float* bo = (const float*)d_in[8];

  char* ws = (char*)d_ws;
  const size_t MB = 1u << 20;
  __bf16* xb  = (__bf16*)(ws + 0 * MB);    // 8 MB x in bf16 [4096,1024]
  __bf16* wqT = (__bf16*)(ws + 8 * MB);    // 2 MB each, transposed bf16 weights
  __bf16* wkT = (__bf16*)(ws + 10 * MB);
  __bf16* wvT = (__bf16*)(ws + 12 * MB);
  __bf16* woT = (__bf16*)(ws + 14 * MB);
  __bf16* Qb  = (__bf16*)(ws + 16 * MB);   // 8 MB [B,H,N,Hd]
  __bf16* Kb  = (__bf16*)(ws + 24 * MB);   // 8 MB [B,H,N,Hd]
  __bf16* Vt  = (__bf16*)(ws + 32 * MB);   // 8 MB [B,H,Hd,N]
  __bf16* Ob  = (__bf16*)(ws + 40 * MB);   // 8 MB [B,N,D]

  k_cvt_bf16<<<(TOKS * DIM) / 256, 256, 0, stream>>>(x, xb, TOKS * DIM);

  dim3 tb(16, 16), tg(DIM / 16, DIM / 16);
  k_transpose_bf16<<<tg, tb, 0, stream>>>(wq, wqT);
  k_transpose_bf16<<<tg, tb, 0, stream>>>(wk, wkT);
  k_transpose_bf16<<<tg, tb, 0, stream>>>(wv, wvT);
  k_transpose_bf16<<<tg, tb, 0, stream>>>(wo, woT);

  k_gemm_qkv<<<dim3(TOKS / 64, DIM / 64, 3), 128, 0, stream>>>(
      xb, wqT, wkT, wvT, bq, bk, bv, Qb, Kb, Vt);

  k_flash_attn<<<(BATCH * NHEAD * (SEQ / 16)) / 4, 128, 0, stream>>>(Qb, Kb, Vt, Ob);

  k_gemm_out<<<dim3(TOKS / 64, DIM / 64), 128, 0, stream>>>(Ob, woT, bo, (float*)d_out);
}
